// BiLSTM_644245094876
// MI455X (gfx1250) — matrix operand order repaired and checked by reading, not compiled, not run on hardware
//
#include <hip/hip_runtime.h>

// ---------------------------------------------------------------------------
// BiLSTM (2 layers, bidirectional) for MI455X / gfx1250.
//   - bf16 WMMA (v_wmma_f32_16x16x32_bf16) for all GEMMs, f32 accumulation
//   - fused input-projection + recurrent GEMM per timestep (weights live in L2)
//   - one kernel launch per timestep per layer; h double-buffered by parity
// ---------------------------------------------------------------------------

typedef __attribute__((ext_vector_type(16))) __bf16       v16bf;
typedef __attribute__((ext_vector_type(8)))  float        v8f;
typedef __attribute__((ext_vector_type(4)))  unsigned int v4u;

#define T_SEQ 512
#define BATCH 64
#define HID   512
#define NGATE 2048   // 4*H
#define IN0   1024   // layer-0 input dim; layer-1 input dim is 2H = 1024 too

__device__ __forceinline__ unsigned short f2bf(float f) {
    unsigned int u = __float_as_uint(f);
    u += 0x7FFFu + ((u >> 16) & 1u);          // round-to-nearest-even
    return (unsigned short)(u >> 16);
}
__device__ __forceinline__ float sigf(float x) { return 1.0f / (1.0f + __expf(-x)); }

union Frag { v16bf v; v4u q[2]; };

__device__ __forceinline__ v16bf ld_frag(const unsigned short* p0,
                                         const unsigned short* p1) {
    Frag f;
    f.q[0] = *(const v4u*)p0;   // global_load_b128
    f.q[1] = *(const v4u*)p1;   // global_load_b128
    return f.v;
}

// ---------------------------------------------------------------------------
__global__ void cvt_f32_bf16(const float* __restrict__ s,
                             unsigned short* __restrict__ d, long n) {
    long i = (long)blockIdx.x * blockDim.x + threadIdx.x;
    long stride = (long)gridDim.x * blockDim.x;
    for (; i < n; i += stride) d[i] = f2bf(s[i]);
}

__global__ void zero_u32(unsigned int* __restrict__ p, long n) {
    long i = (long)blockIdx.x * blockDim.x + threadIdx.x;
    long stride = (long)gridDim.x * blockDim.x;
    for (; i < n; i += stride) p[i] = 0u;
}

// ---------------------------------------------------------------------------
// One LSTM timestep, both directions (blockIdx.z = dir).
// Grid: (8, 1, 2), block: 256 (8 waves).
//   wave = (mw 0..3, ns 0..1): 16 batch rows x 32 hidden cols x 4 gates.
//   Workgroup owns all 64 batch rows x 64 hidden cols x all 4 gates,
//   so the cell update is fully local to the workgroup.
// Fragment layouts follow CDNA5 ISA:
//   A 16x32 bf16: lane m=l&15, elems 0..7 -> K=(l>>4)*8.., elems 8..15 -> K=16+(l>>4)*8..
//   B 32x16 bf16: lane n=l&15, elems 0..7 -> K=(l>>4)*8.., elems 8..15 -> K=16+(l>>4)*8..
//   C/D 16x16 f32: lane col=l&15, elem r -> row=(l>>4)*8+r
// ---------------------------------------------------------------------------
__global__ __launch_bounds__(256)
void lstm_step(const unsigned short* __restrict__ inb,   // [T][B][inK] bf16
               int inK,
               const unsigned short* __restrict__ Wih,   // [2][NGATE][inK] bf16
               const unsigned short* __restrict__ Whh,   // [2][NGATE][HID] bf16
               const float* __restrict__ bih,            // [2][NGATE]
               const float* __restrict__ bhh,            // [2][NGATE]
               unsigned short* __restrict__ hb,          // [2 parity][2 dir][B][HID] bf16
               float* __restrict__ cst,                  // [2 dir][B][HID]
               unsigned short* outb,                     // [T][B][2H] bf16 (layer out)
               float* outf,                              // nullptr, or [T][B][2H] f32
               int s)
{
    const int dir  = blockIdx.z;
    const int t    = dir ? (T_SEQ - 1 - s) : s;
    const int lane = threadIdx.x & 31;
    const int wave = threadIdx.x >> 5;
    const int mw   = wave & 3;          // M-tile (batch)
    const int ns   = wave >> 2;         // N-subtile
    const int m0   = mw * 16;
    const int j0   = blockIdx.x * 64 + ns * 32;   // hidden-unit column base
    const int l15  = lane & 15;
    const int kh   = lane >> 4;

    const unsigned short* Wih_d = Wih + (size_t)dir * NGATE * inK;
    const unsigned short* Whh_d = Whh + (size_t)dir * NGATE * HID;

    const int par = s & 1;
    const unsigned short* hprev = hb + ((size_t)par * 2 + dir) * BATCH * HID;
    unsigned short*       hnext = hb + ((size_t)(1 - par) * 2 + dir) * BATCH * HID;

    // accumulators: acc[gate][coltile] -> 16x16 f32 tile each
    const v8f vzero = {0.f, 0.f, 0.f, 0.f, 0.f, 0.f, 0.f, 0.f};
    v8f acc[4][2];
#pragma unroll
    for (int g = 0; g < 4; ++g)
#pragma unroll
        for (int ct = 0; ct < 2; ++ct) acc[g][ct] = vzero;

    // B-fragment row pointers (one gate row per lane column)
    const unsigned short* bp1[4][2];
    const unsigned short* bp2[4][2];
#pragma unroll
    for (int g = 0; g < 4; ++g)
#pragma unroll
        for (int ct = 0; ct < 2; ++ct) {
            const int n = g * HID + j0 + ct * 16 + l15;
            bp1[g][ct] = Wih_d + (size_t)n * inK + kh * 8;
            bp2[g][ct] = Whh_d + (size_t)n * HID + kh * 8;
        }

    // ---- phase 1: x_t @ W_ih^T ------------------------------------------
    {
        const unsigned short* arow =
            inb + ((size_t)t * BATCH + m0 + l15) * inK + kh * 8;
        for (int k0 = 0; k0 < inK; k0 += 32) {
            v16bf a = ld_frag(arow + k0, arow + k0 + 16);
#pragma unroll
            for (int g = 0; g < 4; ++g)
#pragma unroll
                for (int ct = 0; ct < 2; ++ct) {
                    v16bf b = ld_frag(bp1[g][ct] + k0, bp1[g][ct] + k0 + 16);
                    acc[g][ct] = __builtin_amdgcn_wmma_f32_16x16x32_bf16(
                        false, a, false, b, (short)0, acc[g][ct], false, false);
                }
        }
    }

    // ---- phase 2: h_{t-1} @ W_hh^T --------------------------------------
    {
        const unsigned short* arow = hprev + (size_t)(m0 + l15) * HID + kh * 8;
        for (int k0 = 0; k0 < HID; k0 += 32) {
            v16bf a = ld_frag(arow + k0, arow + k0 + 16);
#pragma unroll
            for (int g = 0; g < 4; ++g)
#pragma unroll
                for (int ct = 0; ct < 2; ++ct) {
                    v16bf b = ld_frag(bp2[g][ct] + k0, bp2[g][ct] + k0 + 16);
                    acc[g][ct] = __builtin_amdgcn_wmma_f32_16x16x32_bf16(
                        false, a, false, b, (short)0, acc[g][ct], false, false);
                }
        }
    }

    // ---- cell update (nonstandard gating, faithful to reference) --------
    //   c_t = c_{t-1}*sig(i) + sig(f)*tanh(g);  h_t = tanh(c_t)*sig(o)
#pragma unroll
    for (int ct = 0; ct < 2; ++ct) {
        const int j = j0 + ct * 16 + l15;
        const float bi = bih[dir * NGATE + 0 * HID + j] + bhh[dir * NGATE + 0 * HID + j];
        const float bf = bih[dir * NGATE + 1 * HID + j] + bhh[dir * NGATE + 1 * HID + j];
        const float bg = bih[dir * NGATE + 2 * HID + j] + bhh[dir * NGATE + 2 * HID + j];
        const float bo = bih[dir * NGATE + 3 * HID + j] + bhh[dir * NGATE + 3 * HID + j];
#pragma unroll
        for (int r = 0; r < 8; ++r) {
            const int b = m0 + kh * 8 + r;
            const float iv = acc[0][ct][r] + bi;
            const float fv = acc[1][ct][r] + bf;
            const float gv = acc[2][ct][r] + bg;
            const float ov = acc[3][ct][r] + bo;

            const size_t cidx = ((size_t)dir * BATCH + b) * HID + j;
            const float cp = cst[cidx];
            const float cn = cp * sigf(iv) + sigf(fv) * tanhf(gv);
            const float hn = tanhf(cn) * sigf(ov);
            cst[cidx] = cn;
            hnext[(size_t)b * HID + j] = f2bf(hn);

            const size_t oidx = ((size_t)t * BATCH + b) * (2 * HID) + dir * HID + j;
            if (outf) outf[oidx] = hn;        // last layer -> f32 d_out
            else      outb[oidx] = f2bf(hn);  // hidden layer -> bf16 for next layer
        }
    }
}

// ---------------------------------------------------------------------------
extern "C" void kernel_launch(void* const* d_in, const int* in_sizes, int n_in,
                              void* d_out, int out_size, void* d_ws, size_t ws_size,
                              hipStream_t stream) {
    (void)in_sizes; (void)n_in; (void)out_size; (void)ws_size;

    const float* x    = (const float*)d_in[0];   // [T,B,IN]
    const float* Wih  = (const float*)d_in[1];   // [2,2,NGATE,IN0]
    const float* bih  = (const float*)d_in[2];   // [2,2,NGATE]
    const float* Whh  = (const float*)d_in[3];   // [2,2,NGATE,HID]
    const float* bhh  = (const float*)d_in[4];   // [2,2,NGATE]
    float*       out  = (float*)d_out;           // [T,B,2H]

    const size_t XE   = (size_t)T_SEQ * BATCH * IN0;      // x / layer-output elems
    const size_t WIHE = (size_t)2 * 2 * NGATE * IN0;
    const size_t WHHE = (size_t)2 * 2 * NGATE * HID;
    const size_t HBE  = (size_t)2 * 2 * BATCH * HID;      // parity x dir x B x H
    const size_t CSE  = (size_t)2 * BATCH * HID;          // dir x B x H

    char* ws = (char*)d_ws;
    size_t off = 0;
    unsigned short* xb    = (unsigned short*)(ws + off); off += XE   * 2;
    unsigned short* out0b = (unsigned short*)(ws + off); off += XE   * 2;
    unsigned short* Wihb  = (unsigned short*)(ws + off); off += WIHE * 2;
    unsigned short* Whhb  = (unsigned short*)(ws + off); off += WHHE * 2;
    unsigned short* hb    = (unsigned short*)(ws + off); off += HBE  * 2;
    float*          cst   = (float*)(ws + off);          off += CSE  * 4;
    // total workspace used: ~153 MB

    // one-time f32 -> bf16 conversions
    cvt_f32_bf16<<<4096, 256, 0, stream>>>(x,   xb,   (long)XE);
    cvt_f32_bf16<<<4096, 256, 0, stream>>>(Wih, Wihb, (long)WIHE);
    cvt_f32_bf16<<<2048, 256, 0, stream>>>(Whh, Whhb, (long)WHHE);

    for (int l = 0; l < 2; ++l) {
        // zero h (both parities, both dirs) and c; hb/cst are adjacent in ws
        zero_u32<<<256, 256, 0, stream>>>((unsigned int*)hb,
                                          (long)((HBE * 2 + CSE * 4) / 4));

        const unsigned short* in  = (l == 0) ? xb : out0b;
        const unsigned short* Wl  = Wihb + (size_t)l * 2 * NGATE * IN0;
        const unsigned short* Whl = Whhb + (size_t)l * 2 * NGATE * HID;
        const float* bil = bih + (size_t)l * 2 * NGATE;
        const float* bhl = bhh + (size_t)l * 2 * NGATE;
        float* of = (l == 1) ? out : nullptr;

        for (int s = 0; s < T_SEQ; ++s) {
            lstm_step<<<dim3(8, 1, 2), 256, 0, stream>>>(
                in, IN0, Wl, Whl, bil, bhl, hb, cst, out0b, of, s);
        }
    }
}
